// BinaryLinear_50285477101619
// MI455X (gfx1250) — compile-verified, matrix operand order checked by reading, not run on hardware
//
#include <hip/hip_runtime.h>
#include <hip/hip_bf16.h>

// ---------------------------------------------------------------------------
// BinaryLinear: y[m,o] = alpha[o] * sum_k x[m,k]*sign(W[o,k]) + bias[o]
// M=8192 (B*S), K=4096, N=4096.  Compute-bound -> bf16 WMMA (sign exact in bf16).
// Wave tile 64x64 (16 accumulators) => 32 FLOP/byte from cache per wave-step.
// ---------------------------------------------------------------------------

#define M_DIM 8192
#define N_DIM 4096
#define K_DIM 4096

typedef __attribute__((ext_vector_type(16))) __bf16    v16bf;
typedef __attribute__((ext_vector_type(8)))  float     v8f;
typedef __attribute__((ext_vector_type(4)))  unsigned  v4u;
typedef __attribute__((ext_vector_type(4)))  float     v4f;
typedef __attribute__((ext_vector_type(4)))  __bf16    v4bf;

union Frag16 {
    v16bf bf;    // WMMA 16-bit A/B operand (8 VGPRs)
    v4u   u[2];  // two 16-byte global loads
};

// ---------------------------------------------------------------------------
// Kernel 1: per-output-row alpha = mean(|W|) and bf16 sign matrix [N][K]
// one 256-thread block per output channel o
// ---------------------------------------------------------------------------
__global__ __launch_bounds__(256) void pack_kernel(
    const float* __restrict__ W,   // [N][K]
    __bf16* __restrict__ S,        // [N][K] bf16 sign
    float* __restrict__ alpha)     // [N]
{
    const int o = blockIdx.x;
    const int t = threadIdx.x;
    const float* row = W + (size_t)o * K_DIM;

    float s = 0.f;
#pragma unroll
    for (int i = 0; i < K_DIM / 256; ++i)
        s += fabsf(row[t + i * 256]);

    __shared__ float red[256];
    red[t] = s;
    __syncthreads();
#pragma unroll
    for (int off = 128; off > 0; off >>= 1) {
        if (t < off) red[t] += red[t + off];
        __syncthreads();
    }
    if (t == 0) alpha[o] = red[0] * (1.0f / (float)K_DIM);

    __bf16* srow = S + (size_t)o * K_DIM;
#pragma unroll
    for (int i = 0; i < K_DIM / 256; ++i) {
        float w  = row[t + i * 256];
        float sg = (w > 0.f) ? 1.f : ((w < 0.f) ? -1.f : 0.f);  // jnp.sign semantics
        srow[t + i * 256] = (__bf16)sg;
    }
}

// ---------------------------------------------------------------------------
// Kernel 2: x f32 -> bf16, 4 elements per thread
// ---------------------------------------------------------------------------
__global__ __launch_bounds__(256) void cvt_kernel(
    const float* __restrict__ x, __bf16* __restrict__ xb, int n4)
{
    int i = blockIdx.x * 256 + threadIdx.x;
    if (i < n4) {
        v4f f = ((const v4f*)x)[i];
        ((v4bf*)xb)[i] = __builtin_convertvector(f, v4bf);
    }
}

// ---------------------------------------------------------------------------
// Kernel 3: WMMA GEMM.
// Block = 8 waves (256 thr) arranged 4(M) x 2(N) -> block tile 256x128.
// Wave  = 64x64: 4 M-tiles x 4 N-tiles of 16x16, K-step 32.
// A operand: x_bf16 [M][K] row-major.  B operand: sign [N][K] row-major.
// Per-lane fragment loads follow the ISA 16-bit operand layout:
//   lanes 0-15 : row = base + (lane&15), K = {k0..k0+7, k0+16..k0+23}
//   lanes 16-31: same row set,           K = {k0+8..+15, k0+24..+31}
// Tile-row displacements are compile-time constants folded into the 24-bit
// instruction offset, so A and B each need only one address register pair.
// ---------------------------------------------------------------------------
__global__ __launch_bounds__(256) void gemm_kernel(
    const __bf16* __restrict__ A,     // [M][K] x in bf16
    const __bf16* __restrict__ Bs,    // [N][K] sign in bf16
    const float*  __restrict__ alpha, // [N]
    const float*  __restrict__ bias,  // [N]
    float* __restrict__ C)            // [M][N]
{
    const int lane = threadIdx.x & 31;
    const int wave = threadIdx.x >> 5;
    const int wm = wave & 3;   // 4 waves over M
    const int wn = wave >> 2;  // 2 waves over N

    const int waveM = blockIdx.y * 256 + wm * 64;  // 4 M-tiles of 16
    const int waveN = blockIdx.x * 128 + wn * 64;  // 4 N-tiles of 16

    const int lr    = lane & 15;
    const int khalf = (lane >> 4) << 3;  // 0 or 8

    const __bf16* aBase = A  + (size_t)(waveM + lr) * K_DIM + khalf;
    const __bf16* bBase = Bs + (size_t)(waveN + lr) * K_DIM + khalf;

    v8f acc[4][4];
#pragma unroll
    for (int i = 0; i < 4; ++i)
#pragma unroll
        for (int j = 0; j < 4; ++j)
#pragma unroll
            for (int e = 0; e < 8; ++e) acc[i][j][e] = 0.f;

    for (int k0 = 0; k0 < K_DIM; k0 += 32) {
        Frag16 a[4], b[4];
#pragma unroll
        for (int t = 0; t < 4; ++t) {
            const __bf16* pa = aBase + (size_t)(t * 16 * K_DIM);  // +t*128KB imm
            a[t].u[0] = *(const v4u*)(pa);
            a[t].u[1] = *(const v4u*)(pa + 16);
            const __bf16* pb = bBase + (size_t)(t * 16 * K_DIM);
            b[t].u[0] = *(const v4u*)(pb);
            b[t].u[1] = *(const v4u*)(pb + 16);
        }
        aBase += 32;
        bBase += 32;

#pragma unroll
        for (int i = 0; i < 4; ++i)
#pragma unroll
            for (int j = 0; j < 4; ++j)
                acc[i][j] = __builtin_amdgcn_wmma_f32_16x16x32_bf16(
                    /*neg_a=*/false, a[i].bf,
                    /*neg_b=*/false, b[j].bf,
                    /*c_mod=*/(short)0, acc[i][j],
                    /*reuse_a=*/false, /*reuse_b=*/false);
    }

    // Epilogue: C/D layout -> VGPR r holds row (r) for lanes 0-15, (r+8) for
    // lanes 16-31; column = lane&15 within each 16-wide N tile.  Row/tile
    // displacements fold into the immediate offset (max ~1MB < 8MB).
    float* cBase = C + (size_t)(waveM + ((lane >> 4) << 3)) * N_DIM + waveN + lr;
#pragma unroll
    for (int j = 0; j < 4; ++j) {
        const int col = waveN + j * 16 + lr;
        const float av = alpha[col];
        const float bv = bias[col];
#pragma unroll
        for (int i = 0; i < 4; ++i) {
#pragma unroll
            for (int r = 0; r < 8; ++r)
                cBase[(size_t)(i * 16 + r) * N_DIM + j * 16] =
                    fmaf(acc[i][j][r], av, bv);
        }
    }
}

// ---------------------------------------------------------------------------
// Launch
// ---------------------------------------------------------------------------
extern "C" void kernel_launch(void* const* d_in, const int* in_sizes, int n_in,
                              void* d_out, int out_size, void* d_ws, size_t ws_size,
                              hipStream_t stream) {
    const float* x    = (const float*)d_in[0];  // [4,2048,4096] = [8192][4096]
    const float* W    = (const float*)d_in[1];  // [4096][4096]
    const float* bias = (const float*)d_in[2];  // [4096]
    float* out = (float*)d_out;                 // [8192][4096]

    char* ws = (char*)d_ws;
    const size_t xbf_bytes  = (size_t)M_DIM * K_DIM * 2;  // 67,108,864
    const size_t sign_bytes = (size_t)N_DIM * K_DIM * 2;  // 33,554,432
    __bf16* x_bf   = (__bf16*)ws;
    __bf16* w_sign = (__bf16*)(ws + xbf_bytes);
    float*  alpha  = (float*)(ws + xbf_bytes + sign_bytes);

    // 1) binarize weights + per-channel alpha
    pack_kernel<<<N_DIM, 256, 0, stream>>>(W, w_sign, alpha);

    // 2) x -> bf16
    const int n4 = (M_DIM * K_DIM) / 4;  // 8,388,608
    cvt_kernel<<<(n4 + 255) / 256, 256, 0, stream>>>(x, x_bf, n4);

    // 3) WMMA GEMM + fused alpha/bias epilogue
    dim3 grid(N_DIM / 128, M_DIM / 256);  // (32, 32); N fastest for L2 reuse
    gemm_kernel<<<grid, 256, 0, stream>>>(x_bf, w_sign, alpha, bias, out);
}
